// MambaBlock_6030134084215
// MI455X (gfx1250) — matrix operand order repaired and checked by reading, not compiled, not run on hardware
//
#include <hip/hip_runtime.h>
#include <hip/hip_bf16.h>

typedef __bf16 bf16;
typedef bf16  v8bf  __attribute__((ext_vector_type(8)));
typedef bf16  v16bf __attribute__((ext_vector_type(16)));
typedef float v8f   __attribute__((ext_vector_type(8)));

#define DIM     1024
#define INNER   2048
#define STATE   16
#define CONVW   4
#define DT_RANK 64
#define BATCH   4
#define SEQ     2048
#define ROWS    (BATCH * SEQ)          // 8192
#define XP_PITCH 128                   // padded (DT_RANK + 2*STATE = 96 -> 128)

// ---- gfx1250 async global->LDS copy (ASYNCcnt path), 16B per lane ----
__device__ __forceinline__ void async_copy16(unsigned lds_off,
                                             unsigned long long gaddr) {
  asm volatile("global_load_async_to_lds_b128 %0, %1, off"
               :: "v"(lds_off), "v"(gaddr) : "memory");
}
__device__ __forceinline__ void async_wait0() {
  asm volatile("s_wait_asynccnt 0" ::: "memory");
}

// ---------------- generic cast f32 -> bf16 ----------------
__global__ void cast_f32_bf16_kernel(const float* __restrict__ in,
                                     bf16* __restrict__ out, long long n) {
  long long i = (long long)blockIdx.x * blockDim.x + threadIdx.x;
  long long stride = (long long)gridDim.x * blockDim.x;
  for (; i < n; i += stride) out[i] = (bf16)in[i];
}

__global__ void fill_zero_bf16_kernel(bf16* __restrict__ p, long long n) {
  long long i = (long long)blockIdx.x * blockDim.x + threadIdx.x;
  if (i < n) p[i] = (bf16)0.0f;
}

// ---------------- bf16 WMMA GEMM: C[M,N] = A[M,K] * B[N,K]^T ----------------
// 256 threads = 8 waves; block tile 128x128; wave tile 64x32 (8 accums);
// K stepped by 32; double-buffered LDS fed by async global->LDS b128 copies.
#define TILE_M 128
#define TILE_N 128
#define TILE_K 32
#define PITCH  40   // bf16 elems; 80B row, 16B aligned, conflict-staggered

__global__ __launch_bounds__(256) void gemm_bf16_wmma_kernel(
    const bf16* __restrict__ A, const bf16* __restrict__ B,
    float* __restrict__ C, int M, int N, int K) {
  __shared__ bf16 Alds[2][TILE_M * PITCH];
  __shared__ bf16 Blds[2][TILE_N * PITCH];

  const int tid   = threadIdx.x;
  const int lane  = tid & 31;
  const int wave  = tid >> 5;
  const int waveM = wave >> 2;             // 0..1 (x64 rows)
  const int waveN = wave & 3;              // 0..3 (x32 cols)
  const size_t bm = (size_t)blockIdx.y * TILE_M;
  const size_t bn = (size_t)blockIdx.x * TILE_N;

  const int mrow  = lane & 15;             // operand M/N index
  const int khalf = lane >> 4;             // K-half select

  // per-thread staged chunks: A and B tiles are each 128 rows x 64B
  // = 512 x 16B chunks; each thread moves 2 A-chunks + 2 B-chunks per K-step.
  int crow[2], cko[2];
#pragma unroll
  for (int i = 0; i < 2; ++i) {
    int c = tid + 256 * i;
    crow[i] = c >> 2;
    cko[i]  = (c & 3) * 8;
  }

  v8f acc[8];
#pragma unroll
  for (int i = 0; i < 8; ++i) acc[i] = (v8f){};

  const int KT = K / TILE_K;

  // prologue: async-stage tile 0 into buffer 0
#pragma unroll
  for (int i = 0; i < 2; ++i) {
    async_copy16((unsigned)(size_t)&Alds[0][crow[i] * PITCH + cko[i]],
                 (unsigned long long)(A + (bm + crow[i]) * (size_t)K + cko[i]));
    async_copy16((unsigned)(size_t)&Blds[0][crow[i] * PITCH + cko[i]],
                 (unsigned long long)(B + (bn + crow[i]) * (size_t)K + cko[i]));
  }
  async_wait0();
  __syncthreads();

  for (int kt = 0; kt < KT; ++kt) {
    const int cur = kt & 1;
    // async-prefetch next K tile into the other buffer while computing
    if (kt + 1 < KT) {
      const int k0 = (kt + 1) * TILE_K;
#pragma unroll
      for (int i = 0; i < 2; ++i) {
        async_copy16((unsigned)(size_t)&Alds[cur ^ 1][crow[i] * PITCH + cko[i]],
                     (unsigned long long)(A + (bm + crow[i]) * (size_t)K + k0 + cko[i]));
        async_copy16((unsigned)(size_t)&Blds[cur ^ 1][crow[i] * PITCH + cko[i]],
                     (unsigned long long)(B + (bn + crow[i]) * (size_t)K + k0 + cko[i]));
      }
    }

    // B operands (32x16 K-major): lane n = lane&15, K-half by lane group.
    union pk { v16bf v; v8bf h[2]; };
    pk bop[2];
#pragma unroll
    for (int u = 0; u < 2; ++u) {
      int base = (waveN * 32 + u * 16 + mrow) * PITCH + khalf * 16;
      bop[u].h[0] = *(const v8bf*)&Blds[cur][base];
      bop[u].h[1] = *(const v8bf*)&Blds[cur][base + 8];
    }
    // A operands (16x32): VGPR0-7 = K khalf*16..+16 (contiguous, matches B).
#pragma unroll
    for (int t = 0; t < 4; ++t) {
      pk aop;
      int base = (waveM * 64 + t * 16 + mrow) * PITCH + khalf * 16;
      aop.h[0] = *(const v8bf*)&Alds[cur][base];
      aop.h[1] = *(const v8bf*)&Alds[cur][base + 8];
#pragma unroll
      for (int u = 0; u < 2; ++u)
        acc[t * 2 + u] = __builtin_amdgcn_wmma_f32_16x16x32_bf16(
            false, aop.v, false, bop[u].v, (short)0, acc[t * 2 + u], false, false);
    }

    async_wait0();     // next tile fully landed in LDS
    __syncthreads();   // all waves done reading current buffer
  }

  // Store: C/D layout — vgpr r: lanes0-15 -> M=r, lanes16-31 -> M=r+8.
  const int nbase = (int)bn + waveN * 32;
  const int nn    = lane & 15;
  const int mb    = (lane >> 4) * 8;
#pragma unroll
  for (int t = 0; t < 4; ++t)
#pragma unroll
    for (int u = 0; u < 2; ++u)
#pragma unroll
      for (int r = 0; r < 8; ++r) {
        size_t m = bm + waveM * 64 + t * 16 + r + mb;
        C[m * (size_t)N + nbase + u * 16 + nn] = acc[t * 2 + u][r];
      }
}

// ---------------- depthwise causal conv4 + SiLU ----------------
__global__ void conv_silu_kernel(const float* __restrict__ xz,
                                 const float* __restrict__ conv_w,
                                 const float* __restrict__ conv_b,
                                 float* __restrict__ xconv_f32,
                                 bf16* __restrict__ xconv_bf16) {
  long long i = (long long)blockIdx.x * blockDim.x + threadIdx.x;
  const long long total = (long long)ROWS * INNER;
  if (i >= total) return;
  int e        = (int)(i & (INNER - 1));
  long long bl = i >> 11;               // b*SEQ + l
  int l        = (int)(bl & (SEQ - 1));

  float acc = conv_b[e];
#pragma unroll
  for (int j = 0; j < CONVW; ++j) {
    int ls = l - (CONVW - 1) + j;
    if (ls >= 0) {
      long long row = bl - (long long)(l - ls);
      acc += conv_w[e * CONVW + j] * xz[row * (2 * INNER) + e];  // x_inner half
    }
  }
  float s = acc / (1.0f + __expf(-acc));   // silu
  xconv_f32[i]  = s;
  xconv_bf16[i] = (bf16)s;
}

// ---------------- extract dt_r (xp[:, :64]) as bf16 ----------------
__global__ void extract_dtr_kernel(const float* __restrict__ xp,
                                   bf16* __restrict__ dtr) {
  long long i = (long long)blockIdx.x * blockDim.x + threadIdx.x;
  const long long n = (long long)ROWS * DT_RANK;
  if (i >= n) return;
  int r       = (int)(i & (DT_RANK - 1));
  long long m = i >> 6;
  dtr[i] = (bf16)xp[m * XP_PITCH + r];
}

// ---------------- selective scan (sequential in L) ----------------
// One thread per (b, e); 16-state recurrence in registers; B_t/C_t in LDS.
__global__ __launch_bounds__(256) void scan_kernel(
    const float* __restrict__ xz,      // [ROWS, 2*INNER] (z in upper half)
    const float* __restrict__ xconv,   // [ROWS, INNER]
    const float* __restrict__ xp,      // [ROWS, XP_PITCH]
    const float* __restrict__ dtraw,   // [ROWS, INNER]
    const float* __restrict__ dt_b,    // [INNER]
    const float* __restrict__ A_log,   // [INNER, STATE]
    const float* __restrict__ Dp,      // [INNER]
    bf16* __restrict__ yout)           // [ROWS, INNER]
{
  __shared__ float bc[2 * STATE];
  const int b = blockIdx.x >> 3;                 // 4 batches x 8 e-blocks
  const int e = (blockIdx.x & 7) * 256 + threadIdx.x;

  float Arow[STATE];
#pragma unroll
  for (int s = 0; s < STATE; ++s) Arow[s] = -__expf(A_log[e * STATE + s]);
  float h[STATE];
#pragma unroll
  for (int s = 0; s < STATE; ++s) h[s] = 0.0f;

  const float bias = dt_b[e];
  const float Dv   = Dp[e];

  for (int l = 0; l < SEQ; ++l) {
    long long row = (long long)b * SEQ + l;
    __syncthreads();
    if (threadIdx.x < 2 * STATE)
      bc[threadIdx.x] = xp[row * XP_PITCH + DT_RANK + threadIdx.x];
    __syncthreads();

    float v  = dtraw[row * INNER + e] + bias;
    float dt = (v > 20.0f) ? v : __logf(1.0f + __expf(v));   // softplus
    float xt = xconv[row * INNER + e];
    float dx = dt * xt;
    float y  = 0.0f;
#pragma unroll
    for (int s = 0; s < STATE; ++s) {
      float dA = __expf(Arow[s] * dt);
      h[s] = dA * h[s] + dx * bc[s];
      y += h[s] * bc[STATE + s];
    }
    y += Dv * xt;
    float z  = xz[row * (2 * INNER) + INNER + e];
    float sz = z / (1.0f + __expf(-z));
    yout[row * INNER + e] = (bf16)(y * sz);
  }
}

// ---------------- host-side launcher ----------------
static inline long long cdiv(long long a, long long b) { return (a + b - 1) / b; }

extern "C" void kernel_launch(void* const* d_in, const int* in_sizes, int n_in,
                              void* d_out, int out_size, void* d_ws, size_t ws_size,
                              hipStream_t stream) {
  const float* x        = (const float*)d_in[0];  // [4,2048,1024]
  const float* in_w     = (const float*)d_in[1];  // [4096,1024]
  const float* conv_w   = (const float*)d_in[2];  // [2048,1,4]
  const float* conv_b   = (const float*)d_in[3];  // [2048]
  const float* xproj_w  = (const float*)d_in[4];  // [96,2048]
  const float* dtproj_w = (const float*)d_in[5];  // [2048,64]
  const float* dtproj_b = (const float*)d_in[6];  // [2048]
  const float* A_log    = (const float*)d_in[7];  // [2048,16]
  const float* Dp       = (const float*)d_in[8];  // [2048]
  const float* out_w    = (const float*)d_in[9];  // [1024,2048]
  float* out            = (float*)d_out;          // [4,2048,1024]

  char* ws = (char*)d_ws;
  size_t off = 0;
  auto carve = [&](size_t bytes) {
    char* p = ws + off;
    off += (bytes + 255) & ~(size_t)255;
    return p;
  };
  bf16*  x_bf      = (bf16*) carve((size_t)ROWS * DIM * 2);
  bf16*  inw_bf    = (bf16*) carve((size_t)2 * INNER * DIM * 2);
  bf16*  xpw_bf    = (bf16*) carve((size_t)XP_PITCH * INNER * 2);  // padded 128 rows
  bf16*  dtw_bf    = (bf16*) carve((size_t)INNER * DT_RANK * 2);
  bf16*  outw_bf   = (bf16*) carve((size_t)DIM * INNER * 2);
  float* xz        = (float*)carve((size_t)ROWS * 2 * INNER * 4);
  float* xconv_f   = (float*)carve((size_t)ROWS * INNER * 4);
  bf16*  xconv_bf  = (bf16*) carve((size_t)ROWS * INNER * 2);
  float* xp        = (float*)carve((size_t)ROWS * XP_PITCH * 4);   // padded pitch
  bf16*  dtr_bf    = (bf16*) carve((size_t)ROWS * DT_RANK * 2);
  float* dtraw     = (float*)carve((size_t)ROWS * INNER * 4);
  bf16*  y_bf      = (bf16*) carve((size_t)ROWS * INNER * 2);
  (void)ws_size;

  const int T = 256;
  auto cast = [&](const float* src, bf16* dst, long long n) {
    int grid = (int)cdiv(n, T);
    cast_f32_bf16_kernel<<<grid, T, 0, stream>>>(src, dst, n);
  };

  // 1) casts to bf16 (x_proj weight zero-padded to 128 rows)
  cast(x,        x_bf,    (long long)ROWS * DIM);
  cast(in_w,     inw_bf,  (long long)2 * INNER * DIM);
  {
    long long n = (long long)XP_PITCH * INNER;
    fill_zero_bf16_kernel<<<(int)cdiv(n, T), T, 0, stream>>>(xpw_bf, n);
  }
  cast(xproj_w,  xpw_bf,  (long long)(DT_RANK + 2 * STATE) * INNER);
  cast(dtproj_w, dtw_bf,  (long long)INNER * DT_RANK);
  cast(out_w,    outw_bf, (long long)DIM * INNER);

  // 2) in_proj: xz[8192,4096] = x[8192,1024] @ in_w[4096,1024]^T
  {
    dim3 g(2 * INNER / TILE_N, ROWS / TILE_M);
    gemm_bf16_wmma_kernel<<<g, 256, 0, stream>>>(x_bf, inw_bf, xz,
                                                 ROWS, 2 * INNER, DIM);
  }
  // 3) conv + silu
  {
    long long n = (long long)ROWS * INNER;
    conv_silu_kernel<<<(int)cdiv(n, T), T, 0, stream>>>(xz, conv_w, conv_b,
                                                        xconv_f, xconv_bf);
  }
  // 4) x_proj (padded): xp[8192,128] = xconv[8192,2048] @ xpw_pad[128,2048]^T
  {
    dim3 g(XP_PITCH / TILE_N, ROWS / TILE_M);
    gemm_bf16_wmma_kernel<<<g, 256, 0, stream>>>(xconv_bf, xpw_bf, xp,
                                                 ROWS, XP_PITCH, INNER);
  }
  // 5) dt_r extraction + dt_proj GEMM: dtraw[8192,2048]
  {
    long long n = (long long)ROWS * DT_RANK;
    extract_dtr_kernel<<<(int)cdiv(n, T), T, 0, stream>>>(xp, dtr_bf);
    dim3 g(INNER / TILE_N, ROWS / TILE_M);
    gemm_bf16_wmma_kernel<<<g, 256, 0, stream>>>(dtr_bf, dtw_bf, dtraw,
                                                 ROWS, INNER, DT_RANK);
  }
  // 6) selective scan -> y (bf16), with D*x and silu(z) gating fused
  {
    dim3 g(BATCH * (INNER / 256));
    scan_kernel<<<g, 256, 0, stream>>>(xz, xconv_f, xp, dtraw,
                                       dtproj_b, A_log, Dp, y_bf);
  }
  // 7) out_proj: out[8192,1024] = y[8192,2048] @ out_w[1024,2048]^T
  {
    dim3 g(DIM / TILE_N, ROWS / TILE_M);
    gemm_bf16_wmma_kernel<<<g, 256, 0, stream>>>(y_bf, outw_bf, out,
                                                 ROWS, DIM, INNER);
  }
}
